// _Attention_7310034337795
// MI455X (gfx1250) — compile-verified, matrix operand order checked by reading, not run on hardware
//
#include <hip/hip_runtime.h>
#include <hip/hip_bf16.h>

// ---------------------------------------------------------------------------
// Attention forward on gfx1250: x@Wqkv -> flash attention -> @Wproj + b
// All matmuls via v_wmma_f32_16x16x32_bf16 (bf16 operands, f32 accumulate).
// K tiles staged with GLOBAL_LOAD_ASYNC_TO_LDS when the builtin is available.
// ---------------------------------------------------------------------------

typedef __bf16 bf16_t;
typedef __bf16 v8bf  __attribute__((ext_vector_type(8)));
typedef __bf16 v16bf __attribute__((ext_vector_type(16)));
typedef float  v8f   __attribute__((ext_vector_type(8)));
typedef float  v4f   __attribute__((ext_vector_type(4)));
typedef int    v4i   __attribute__((ext_vector_type(4)));

#define AS_GLOBAL __attribute__((address_space(1)))
#define AS_LDS    __attribute__((address_space(3)))

#define NTOK 4096
#define CDIM 768
#define NH   12
#define HD   64
#define QKVC (3 * CDIM)          // 2304
#define ATT_SCALE 0.125f         // HD^-0.5

#if defined(__gfx1250__) && __has_builtin(__builtin_amdgcn_global_load_async_to_lds_b128)
#define USE_ASYNC_LDS 1
#else
#define USE_ASYNC_LDS 0
#endif

// concat two 8-elem halves into the 16-elem WMMA fragment
// (elements 0..7 = K kb..kb+7, elements 8..15 = K kb+16..kb+23 per ISA layout)
static __device__ __forceinline__ v16bf cat8(v8bf lo, v8bf hi) {
    v16bf r;
#pragma unroll
    for (int i = 0; i < 8; ++i) { r[i] = lo[i]; r[8 + i] = hi[i]; }
    return r;
}

static __device__ __forceinline__ float redmax16(float v) {
    // row-reduce across the 16-lane half-groups (masks < 16 stay in-group)
    v = fmaxf(v, __shfl_xor(v, 1));
    v = fmaxf(v, __shfl_xor(v, 2));
    v = fmaxf(v, __shfl_xor(v, 4));
    v = fmaxf(v, __shfl_xor(v, 8));
    return v;
}
static __device__ __forceinline__ float redsum16(float v) {
    v += __shfl_xor(v, 1);
    v += __shfl_xor(v, 2);
    v += __shfl_xor(v, 4);
    v += __shfl_xor(v, 8);
    return v;
}

// ---------------------------------------------------------------------------
// Generic tiled GEMM: C[M,N] = A[M,K] * B[K,N] (+bias), A/B f32 -> bf16 WMMA.
// Block tile 128x128, K-tile 32, 256 threads = 8 waves in a 4x2 grid;
// each wave owns a 32x64 sub-tile = 2x4 WMMA accumulators.
// Next K-chunk is prefetched to VGPRs so global latency overlaps the WMMAs.
// ---------------------------------------------------------------------------
template <typename OutT, bool BIAS>
__global__ __launch_bounds__(256) void gemm_wmma_bf16(
    const float* __restrict__ A, const float* __restrict__ B,
    const float* __restrict__ bias, OutT* __restrict__ C,
    int M, int N, int K)
{
    __shared__ bf16_t As[128][40];   // [row][k]   (pad 32->40, 80B stride, 16B aligned)
    __shared__ bf16_t Bs[128][40];   // [col][k]   transposed B tile

    const int tid  = threadIdx.x;
    const int lane = tid & 31;
    const int wid  = tid >> 5;
    const int wm   = wid >> 1;            // 0..3
    const int wn   = wid & 1;             // 0..1
    const int m0   = blockIdx.y * 128;
    const int n0   = blockIdx.x * 128;
    const int nl   = lane & 15;           // M (A) / N (B,C) index within tile
    const int kb   = (lane >> 4) * 8;     // K-half base per ISA fragment layout

    v8f acc[2][4];
#pragma unroll
    for (int a = 0; a < 2; ++a)
#pragma unroll
        for (int b = 0; b < 4; ++b)
#pragma unroll
            for (int e = 0; e < 8; ++e) acc[a][b][e] = 0.0f;

    v4f ra[4], rb[4];
    // prologue: prefetch K-chunk 0
#pragma unroll
    for (int it = 0; it < 4; ++it) {
        int slot = tid + it * 256;                    // A: 1024 float4 slots
        int r    = slot >> 3;
        int c4   = (slot & 7) * 4;
        ra[it] = *(const v4f*)&A[(size_t)(m0 + r) * K + c4];
    }
#pragma unroll
    for (int it = 0; it < 4; ++it) {
        int slot = tid + it * 256;                    // B: 1024 float4 slots
        int kr   = slot >> 5;
        int c4   = (slot & 31) * 4;
        rb[it] = *(const v4f*)&B[(size_t)kr * N + n0 + c4];
    }

    for (int kc = 0; kc < K; kc += 32) {
        // stage prefetched registers -> LDS (bf16)
#pragma unroll
        for (int it = 0; it < 4; ++it) {
            int slot = tid + it * 256;
            int r    = slot >> 3;
            int c4   = (slot & 7) * 4;
#pragma unroll
            for (int j = 0; j < 4; ++j) As[r][c4 + j] = (bf16_t)ra[it][j];
        }
#pragma unroll
        for (int it = 0; it < 4; ++it) {
            int slot = tid + it * 256;
            int kr   = slot >> 5;
            int c4   = (slot & 31) * 4;
#pragma unroll
            for (int j = 0; j < 4; ++j) Bs[c4 + j][kr] = (bf16_t)rb[it][j];
        }
        __syncthreads();

        // prefetch next K-chunk while WMMAs run
        if (kc + 32 < K) {
#pragma unroll
            for (int it = 0; it < 4; ++it) {
                int slot = tid + it * 256;
                int r    = slot >> 3;
                int c4   = (slot & 7) * 4;
                ra[it] = *(const v4f*)&A[(size_t)(m0 + r) * K + kc + 32 + c4];
            }
#pragma unroll
            for (int it = 0; it < 4; ++it) {
                int slot = tid + it * 256;
                int kr   = slot >> 5;
                int c4   = (slot & 31) * 4;
                rb[it] = *(const v4f*)&B[(size_t)(kc + 32 + kr) * N + n0 + c4];
            }
        }

        v16bf af[2], bf_[4];
#pragma unroll
        for (int tm = 0; tm < 2; ++tm) {
            int r = wm * 32 + tm * 16 + nl;
            af[tm] = cat8(*(const v8bf*)&As[r][kb], *(const v8bf*)&As[r][kb + 16]);
        }
#pragma unroll
        for (int tn = 0; tn < 4; ++tn) {
            int c = wn * 64 + tn * 16 + nl;
            bf_[tn] = cat8(*(const v8bf*)&Bs[c][kb], *(const v8bf*)&Bs[c][kb + 16]);
        }
#pragma unroll
        for (int tm = 0; tm < 2; ++tm)
#pragma unroll
            for (int tn = 0; tn < 4; ++tn)
                acc[tm][tn] = __builtin_amdgcn_wmma_f32_16x16x32_bf16(
                    false, af[tm], false, bf_[tn], (short)0, acc[tm][tn],
                    false, false);
        __syncthreads();   // LDS free for next stage
    }

    // store: C/D layout -> lane = col, VGPR i = row i (+8 for upper half-wave)
#pragma unroll
    for (int tm = 0; tm < 2; ++tm) {
#pragma unroll
        for (int tn = 0; tn < 4; ++tn) {
            int   col   = n0 + wn * 64 + tn * 16 + nl;
            float bv    = BIAS ? bias[col] : 0.0f;
            int   rbase = m0 + wm * 32 + tm * 16 + ((lane >> 4) * 8);
#pragma unroll
            for (int i = 0; i < 8; ++i) {
                float v = acc[tm][tn][i] + bv;
                C[(size_t)(rbase + i) * N + col] = (OutT)v;
            }
        }
    }
}

// ---------------------------------------------------------------------------
// Head-major V transpose: qkv V-third [token][h*64+d] -> vt[(h*64+d)][token]
// so attention PV B-fragments become contiguous 16B global loads.
// ---------------------------------------------------------------------------
__global__ __launch_bounds__(256) void transpose_v(
    const bf16_t* __restrict__ qkv, bf16_t* __restrict__ vt)
{
    __shared__ bf16_t T[64][72];
    const int tid = threadIdx.x;
    const int t0  = blockIdx.x * 64;   // token tile
    const int h   = blockIdx.y;

#pragma unroll
    for (int it = 0; it < 2; ++it) {
        int slot = tid + it * 256;     // 512 v8 chunks
        int r    = slot >> 3;          // token row 0..63
        int c    = (slot & 7) * 8;     // dim
        *(v8bf*)&T[r][c] =
            *(const v8bf*)&qkv[(size_t)(t0 + r) * QKVC + 2 * CDIM + h * HD + c];
    }
    __syncthreads();
#pragma unroll
    for (int it = 0; it < 2; ++it) {
        int slot = tid + it * 256;
        int d    = slot >> 3;          // dim 0..63
        int c    = (slot & 7) * 8;     // token offset
        v8bf v;
#pragma unroll
        for (int j = 0; j < 8; ++j) v[j] = T[c + j][d];
        *(v8bf*)&vt[(size_t)(h * HD + d) * NTOK + t0 + c] = v;
    }
}

// ---------------------------------------------------------------------------
// Flash attention: one block = 8 waves = 128 query rows of one head.
// K tiles (32 keys x 64 dims) staged to LDS (async-to-LDS when available);
// V fragments loaded straight from the head-major transposed copy;
// online softmax in registers; P round-trips wave-private LDS (no WG barrier).
// ---------------------------------------------------------------------------
__global__ __launch_bounds__(256) void flash_attn_wmma(
    const bf16_t* __restrict__ qkv, const bf16_t* __restrict__ vt,
    float* __restrict__ out)
{
    __shared__ bf16_t Kt[32][72];        // [key][dim]  (pad 64->72)
    __shared__ bf16_t Pt[8][16][32];     // per-wave P scratch [row][key]

    const int tid  = threadIdx.x;
    const int lane = tid & 31;
    const int w    = tid >> 5;           // wave 0..7
    const int h    = blockIdx.y;
    const int q0   = blockIdx.x * 128 + w * 16;   // this wave's query rows
    const int nl   = lane & 15;
    const int kb   = (lane >> 4) * 8;
    const int hoff = h * HD;

    // Q fragments straight from global (bf16, fragment runs are contiguous),
    // pre-scaled by HD^-0.5 (exact in bf16: power of two).
    v16bf qa[2];
#pragma unroll
    for (int c = 0; c < 2; ++c) {
        const bf16_t* p = qkv + (size_t)(q0 + nl) * QKVC + hoff + c * 32 + kb;
        v16bf f = cat8(*(const v8bf*)p, *(const v8bf*)(p + 16));
#pragma unroll
        for (int i = 0; i < 16; ++i) f[i] = (bf16_t)((float)f[i] * ATT_SCALE);
        qa[c] = f;
    }

    float mrow[8], lsum[8];
    v8f   O[4];
#pragma unroll
    for (int i = 0; i < 8; ++i) { mrow[i] = -1.0e30f; lsum[i] = 0.0f; }
#pragma unroll
    for (int j = 0; j < 4; ++j)
#pragma unroll
        for (int e = 0; e < 8; ++e) O[j][e] = 0.0f;

    // fixed per-thread K-staging chunk: 256 threads x 16B = 4KB tile
    const int skey  = tid >> 3;
    const int spart = (tid & 7) * 8;

    for (int kt = 0; kt < NTOK / 32; ++kt) {
        __syncthreads();   // everyone done reading previous Kt
        {
            const bf16_t* gp =
                &qkv[(size_t)(kt * 32 + skey) * QKVC + CDIM + hoff + spart];
#if USE_ASYNC_LDS
            // builtin expects (int4 addrspace(1)*, int4 addrspace(3)*, imm, imm)
            __builtin_amdgcn_global_load_async_to_lds_b128(
                (AS_GLOBAL v4i*)gp,
                (AS_LDS    v4i*)&Kt[skey][spart], 0, 0);
#if __has_builtin(__builtin_amdgcn_s_wait_asynccnt)
            __builtin_amdgcn_s_wait_asynccnt(0);
#else
            asm volatile("s_wait_asynccnt 0" ::: "memory");
#endif
#else
            *(v8bf*)&Kt[skey][spart] = *(const v8bf*)gp;
#endif
        }
        __syncthreads();

        // prefetch V fragments from head-major transposed copy (contiguous 16B)
        v16bf vf[4];
#pragma unroll
        for (int j = 0; j < 4; ++j) {
            const bf16_t* vp =
                &vt[(size_t)(hoff + j * 16 + nl) * NTOK + kt * 32 + kb];
            vf[j] = cat8(*(const v8bf*)vp, *(const v8bf*)(vp + 16));
        }

        // S = Q K^T for 32 keys: two 16-key column groups, K-dim 64 = 2 WMMAs
        v8f S0, S1;
#pragma unroll
        for (int e = 0; e < 8; ++e) { S0[e] = 0.0f; S1[e] = 0.0f; }
#pragma unroll
        for (int c = 0; c < 2; ++c) {
            v16bf b0 = cat8(*(const v8bf*)&Kt[nl][c * 32 + kb],
                            *(const v8bf*)&Kt[nl][c * 32 + kb + 16]);
            v16bf b1 = cat8(*(const v8bf*)&Kt[16 + nl][c * 32 + kb],
                            *(const v8bf*)&Kt[16 + nl][c * 32 + kb + 16]);
            S0 = __builtin_amdgcn_wmma_f32_16x16x32_bf16(false, qa[c], false, b0,
                                                         (short)0, S0, false, false);
            S1 = __builtin_amdgcn_wmma_f32_16x16x32_bf16(false, qa[c], false, b1,
                                                         (short)0, S1, false, false);
        }

        // online softmax (per VGPR i = one row of this half-wave's 8 rows)
#pragma unroll
        for (int i = 0; i < 8; ++i) {
            float t  = redmax16(fmaxf(S0[i], S1[i]));
            float mn = fmaxf(mrow[i], t);
            float al = __expf(mrow[i] - mn);
            float p0 = __expf(S0[i] - mn);
            float p1 = __expf(S1[i] - mn);
            lsum[i]  = lsum[i] * al + redsum16(p0 + p1);
            mrow[i]  = mn;
#pragma unroll
            for (int j = 0; j < 4; ++j) O[j][i] *= al;
            S0[i] = p0;
            S1[i] = p1;
        }

        // spill P (C/D layout) to wave-private LDS, reload as A fragment.
        // Same-wave DS ops execute in order; s_wait_dscnt + memory clobber
        // keeps the compiler from reordering -- no workgroup barrier needed.
        {
            int rb = (lane >> 4) * 8;
#pragma unroll
            for (int i = 0; i < 8; ++i) {
                Pt[w][rb + i][nl]      = (bf16_t)S0[i];
                Pt[w][rb + i][16 + nl] = (bf16_t)S1[i];
            }
        }
        asm volatile("s_wait_dscnt 0" ::: "memory");

        v16bf pf = cat8(*(const v8bf*)&Pt[w][nl][kb],
                        *(const v8bf*)&Pt[w][nl][kb + 16]);
#pragma unroll
        for (int j = 0; j < 4; ++j)
            O[j] = __builtin_amdgcn_wmma_f32_16x16x32_bf16(false, pf, false, vf[j],
                                                           (short)0, O[j], false, false);
    }

    // normalize and store f32 attention output (N, C) with head offset
#pragma unroll
    for (int i = 0; i < 8; ++i) {
        float inv = 1.0f / lsum[i];
        int   row = q0 + (lane >> 4) * 8 + i;
#pragma unroll
        for (int j = 0; j < 4; ++j)
            out[(size_t)row * CDIM + hoff + j * 16 + nl] = O[j][i] * inv;
    }
}

// ---------------------------------------------------------------------------
extern "C" void kernel_launch(void* const* d_in, const int* in_sizes, int n_in,
                              void* d_out, int out_size, void* d_ws, size_t ws_size,
                              hipStream_t stream) {
    (void)in_sizes; (void)n_in; (void)out_size; (void)ws_size;
    const float* x      = (const float*)d_in[0];
    const float* w_qkv  = (const float*)d_in[1];
    const float* w_proj = (const float*)d_in[2];
    const float* b_proj = (const float*)d_in[3];
    float*       out    = (float*)d_out;

    // workspace layout:
    //   qkv  bf16 [4096][2304]            18,874,368 B
    //   vt   bf16 [768][4096] (head-major) 6,291,456 B
    //   attn f32  [4096][768]             12,582,912 B
    bf16_t* qkv  = (bf16_t*)d_ws;
    bf16_t* vt   = (bf16_t*)((char*)d_ws + (size_t)NTOK * QKVC * sizeof(bf16_t));
    float*  attn = (float*)((char*)vt + (size_t)CDIM * NTOK * sizeof(bf16_t));

    // 1) QKV projection -> bf16
    gemm_wmma_bf16<bf16_t, false>
        <<<dim3(QKVC / 128, NTOK / 128), 256, 0, stream>>>(
            x, w_qkv, nullptr, qkv, NTOK, QKVC, CDIM);

    // 2) head-major transpose of V
    transpose_v<<<dim3(NTOK / 64, NH), 256, 0, stream>>>(qkv, vt);

    // 3) flash attention per (query-block, head)
    flash_attn_wmma<<<dim3(NTOK / 128, NH), 256, 0, stream>>>(qkv, vt, attn);

    // 4) output projection + bias -> f32
    gemm_wmma_bf16<float, true>
        <<<dim3(CDIM / 128, NTOK / 128), 256, 0, stream>>>(
            attn, w_proj, b_proj, out, NTOK, CDIM, CDIM);
}
